// GCN_31207232372931
// MI455X (gfx1250) — matrix operand order repaired and checked by reading, not compiled, not run on hardware
//
#include <hip/hip_runtime.h>
#include <hip/hip_bf16.h>

// ---------------------------------------------------------------------------
// CDNA5 (gfx1250) GCN forward: 3 x (WMMA bf16 GEMM -> normalized aggregation)
// ---------------------------------------------------------------------------

typedef __attribute__((ext_vector_type(16))) __bf16 v16bf;
typedef __attribute__((ext_vector_type(8)))  float  v8f;

__device__ __forceinline__ unsigned short f2bf(float f) {
    unsigned int u = __float_as_uint(f);
    // round-to-nearest-even to bf16
    unsigned int r = (u + 0x7FFFu + ((u >> 16) & 1u)) >> 16;
    return (unsigned short)r;
}

// ---- float32 -> bf16 conversion (4 elems/thread) --------------------------
__global__ void cvt_bf16_kernel(const float* __restrict__ in,
                                unsigned short* __restrict__ out, int n4) {
    int t = blockIdx.x * blockDim.x + threadIdx.x;
    if (t >= n4) return;
    float4 v = ((const float4*)in)[t];
    unsigned short* o = out + t * 4;
    o[0] = f2bf(v.x); o[1] = f2bf(v.y); o[2] = f2bf(v.z); o[3] = f2bf(v.w);
}

// ---- degree / dis ----------------------------------------------------------
__global__ void fill_deg_kernel(float* __restrict__ deg, int n) {
    int t = blockIdx.x * blockDim.x + threadIdx.x;
    if (t < n) deg[t] = 1.0f;  // self-loop
}
__global__ void edge_deg_kernel(const int* __restrict__ dst,
                                float* __restrict__ deg, int e) {
    int t = blockIdx.x * blockDim.x + threadIdx.x;
    if (t < e) atomicAdd(&deg[dst[t]], 1.0f);
}
__global__ void rsqrt_kernel(float* __restrict__ deg, int n) {
    int t = blockIdx.x * blockDim.x + threadIdx.x;
    if (t < n) deg[t] = rsqrtf(deg[t]);  // deg >= 1 always
}

// ---- pack W[K, Ncols] (fp32, row-major) into B fragments, contiguous k order (k = 16h + s) --
// B (32x16 bf16) per k-step: lane L -> col n = L&15, half h = L>>4,
// dword v (0..7) holds K = kk*32 + h*16 + 2v (lo) and +1 (hi).
// Storage: dword index = ((c*KT + kk)*32 + lane)*8 + v   (KT = K/32)
__global__ void pack_w_kernel(const float* __restrict__ W,
                              unsigned short* __restrict__ Bp,
                              int K, int Ncols, int NcolTiles) {
    int idx = blockIdx.x * blockDim.x + threadIdx.x;
    int KT = K >> 5;
    int total = NcolTiles * KT * 32 * 8;
    if (idx >= total) return;
    int v    = idx & 7;
    int lane = (idx >> 3) & 31;
    int kk   = (idx >> 8) % KT;
    int c    = idx / (KT * 256);
    int n  = c * 16 + (lane & 15);
    int h  = lane >> 4;
    int k0 = kk * 32 + h * 16 + 2 * v;
    float lo = 0.f, hi = 0.f;
    if (n < Ncols) { lo = W[k0 * Ncols + n]; hi = W[(k0 + 1) * Ncols + n]; }
    Bp[idx * 2 + 0] = f2bf(lo);
    Bp[idx * 2 + 1] = f2bf(hi);
}

// ---- WMMA bf16 GEMM: C[M, ncols] = A[M,128] * W ---------------------------
// One wave (32 threads) per 16x16 C tile; K = 128 = 4 steps of 32.
__global__ void __launch_bounds__(32)
gemm_wmma_bf16_kernel(const unsigned short* __restrict__ A,  // [M,128] bf16
                      const unsigned short* __restrict__ Bp, // packed frags
                      float* __restrict__ C, int ldc, int ncols) {
    const int lane    = threadIdx.x;
    const int rowTile = blockIdx.x;
    const int colTile = blockIdx.y;
    const int r = lane & 15;
    const int h = lane >> 4;
    const int row = rowTile * 16 + r;

    union Frag { v16bf v; uint4 q[2]; } a, b;
    v8f acc = {};

    const unsigned short* Arow = A + (size_t)row * 128;
    const unsigned int*   Bq   = (const unsigned int*)Bp +
                                 ((size_t)(colTile * 4) * 32 + lane) * 8;
#pragma unroll
    for (int kk = 0; kk < 4; ++kk) {
        // A fragment: lane holds K in [kk*32+h*16, +16)
        int k0 = kk * 32 + h * 16;
        a.q[0] = *(const uint4*)(Arow + k0);
        a.q[1] = *(const uint4*)(Arow + k0 + 8);
        const uint4* bq = (const uint4*)(Bq + (size_t)kk * 32 * 8);
        b.q[0] = bq[0];
        b.q[1] = bq[1];
        acc = __builtin_amdgcn_wmma_f32_16x16x32_bf16(
            false, a.v, false, b.v, (short)0, acc, false, false);
    }
    // C layout: VGPR v, lane L -> row M = v + 8*h, col N = L&15
    int col = colTile * 16 + r;
    if (col < ncols) {
#pragma unroll
        for (int v = 0; v < 8; ++v) {
            int m = rowTile * 16 + v + 8 * h;
            C[(size_t)m * ldc + col] = acc[v];
        }
    }
}

// ---- self-loop init: out[i,:] = dis[i]^2 * h[i,:] + b ----------------------
__global__ void self_init_kernel(const float* __restrict__ H,
                                 const float* __restrict__ bias,
                                 const float* __restrict__ dis,
                                 float* __restrict__ out, int F4, int total) {
    int t = blockIdx.x * blockDim.x + threadIdx.x;
    if (t >= total) return;
    int i = t / F4, j = t - i * F4;
    float s  = dis[i];
    float n2 = s * s;
    float4 hv = ((const float4*)H)[(size_t)i * F4 + j];
    float4 bv = ((const float4*)bias)[j];
    float4 o;
    o.x = fmaf(n2, hv.x, bv.x); o.y = fmaf(n2, hv.y, bv.y);
    o.z = fmaf(n2, hv.z, bv.z); o.w = fmaf(n2, hv.w, bv.w);
    ((float4*)out)[(size_t)i * F4 + j] = o;
}

// ---- edge scatter: out[dst,:] += dis[src]*dis[dst] * h[src,:] --------------
__global__ void edge_agg_kernel(const float* __restrict__ H,
                                const float* __restrict__ dis,
                                const int* __restrict__ src,
                                const int* __restrict__ dst,
                                float* __restrict__ out, int F4, int total) {
    int t = blockIdx.x * blockDim.x + threadIdx.x;
    if (t >= total) return;
    int e = t / F4, j = t - e * F4;
    int s = src[e], d = dst[e];
    float nm = dis[s] * dis[d];
    float4 hv = ((const float4*)H)[(size_t)s * F4 + j];
    float* o = out + ((size_t)d * F4 + j) * 4;
    atomicAdd(o + 0, nm * hv.x);
    atomicAdd(o + 1, nm * hv.y);
    atomicAdd(o + 2, nm * hv.z);
    atomicAdd(o + 3, nm * hv.w);
}

// ---- relu + reconvert to bf16 for the next layer's GEMM A ------------------
__global__ void relu_cvt_kernel(const float* __restrict__ in,
                                unsigned short* __restrict__ out, int n4) {
    int t = blockIdx.x * blockDim.x + threadIdx.x;
    if (t >= n4) return;
    float4 v = ((const float4*)in)[t];
    unsigned short* o = out + t * 4;
    o[0] = f2bf(fmaxf(v.x, 0.f)); o[1] = f2bf(fmaxf(v.y, 0.f));
    o[2] = f2bf(fmaxf(v.z, 0.f)); o[3] = f2bf(fmaxf(v.w, 0.f));
}

// ---------------------------------------------------------------------------
static inline size_t alignup(size_t x) { return (x + 255) & ~(size_t)255; }
static inline int cdiv(int a, int b) { return (a + b - 1) / b; }

extern "C" void kernel_launch(void* const* d_in, const int* in_sizes, int n_in,
                              void* d_out, int out_size, void* d_ws, size_t ws_size,
                              hipStream_t stream) {
    const float* x  = (const float*)d_in[0];
    const int*   ei = (const int*)d_in[1];
    const float* W1 = (const float*)d_in[2];
    const float* b1 = (const float*)d_in[3];
    const float* W2 = (const float*)d_in[4];
    const float* b2 = (const float*)d_in[5];
    const float* W3 = (const float*)d_in[6];
    const float* b3 = (const float*)d_in[7];

    const int F  = 128;                 // D_FEAT == HIDDEN == 128
    const int N  = in_sizes[0] / F;     // 50000
    const int E  = in_sizes[1] / 2;     // 800000
    const int C3 = in_sizes[7];         // 40 classes
    const int* src = ei;
    const int* dst = ei + E;
    const int ct3 = cdiv(C3, 16);       // 3 col tiles, masked store

    // ---- workspace layout ----
    char* ws = (char*)d_ws;
    size_t o = 0;
    float* dis = (float*)(ws + o);            o += alignup((size_t)N * 4);
    unsigned short* Abf = (unsigned short*)(ws + o); o += alignup((size_t)N * F * 2);
    float* H   = (float*)(ws + o);            o += alignup((size_t)N * F * 4);
    float* AGG = (float*)(ws + o);            o += alignup((size_t)N * F * 4);
    unsigned short* W1p = (unsigned short*)(ws + o); o += alignup((size_t)8   * 4 * 32 * 8 * 2 * 2);
    unsigned short* W2p = (unsigned short*)(ws + o); o += alignup((size_t)8   * 4 * 32 * 8 * 2 * 2);
    unsigned short* W3p = (unsigned short*)(ws + o); o += alignup((size_t)ct3 * 4 * 32 * 8 * 2 * 2);
    (void)ws_size; (void)n_in; (void)out_size;

    const int BLK = 256;

    // ---- prep: bf16 features, degrees, packed weights ----
    int nx4 = N * F / 4;
    cvt_bf16_kernel<<<cdiv(nx4, BLK), BLK, 0, stream>>>(x, Abf, nx4);
    fill_deg_kernel<<<cdiv(N, BLK), BLK, 0, stream>>>(dis, N);
    edge_deg_kernel<<<cdiv(E, BLK), BLK, 0, stream>>>(dst, dis, E);
    rsqrt_kernel<<<cdiv(N, BLK), BLK, 0, stream>>>(dis, N);

    pack_w_kernel<<<cdiv(8 * 4 * 256, BLK), BLK, 0, stream>>>(W1, W1p, F, F, 8);
    pack_w_kernel<<<cdiv(8 * 4 * 256, BLK), BLK, 0, stream>>>(W2, W2p, F, F, 8);
    pack_w_kernel<<<cdiv(ct3 * 4 * 256, BLK), BLK, 0, stream>>>(W3, W3p, F, C3, ct3);

    const int rowTiles = N / 16;      // 3125 (N divisible by 16)
    const int F4  = F / 4;
    const int C34 = C3 / 4;

    // ================= layer 1 =================
    {
        dim3 g(rowTiles, 8);
        gemm_wmma_bf16_kernel<<<g, 32, 0, stream>>>(Abf, W1p, H, F, F);
        int ts = N * F4;
        self_init_kernel<<<cdiv(ts, BLK), BLK, 0, stream>>>(H, b1, dis, AGG, F4, ts);
        int te = E * F4;
        edge_agg_kernel<<<cdiv(te, BLK), BLK, 0, stream>>>(H, dis, src, dst, AGG, F4, te);
        relu_cvt_kernel<<<cdiv(ts, BLK), BLK, 0, stream>>>(AGG, Abf, ts);
    }
    // ================= layer 2 =================
    {
        dim3 g(rowTiles, 8);
        gemm_wmma_bf16_kernel<<<g, 32, 0, stream>>>(Abf, W2p, H, F, F);
        int ts = N * F4;
        self_init_kernel<<<cdiv(ts, BLK), BLK, 0, stream>>>(H, b2, dis, AGG, F4, ts);
        int te = E * F4;
        edge_agg_kernel<<<cdiv(te, BLK), BLK, 0, stream>>>(H, dis, src, dst, AGG, F4, te);
        relu_cvt_kernel<<<cdiv(ts, BLK), BLK, 0, stream>>>(AGG, Abf, ts);
    }
    // ================= layer 3 (output, 40 cols) =================
    {
        float* OUT = (float*)d_out;
        dim3 g(rowTiles, ct3);
        gemm_wmma_bf16_kernel<<<g, 32, 0, stream>>>(Abf, W3p, H, C3, C3);
        int ts = N * C34;
        self_init_kernel<<<cdiv(ts, BLK), BLK, 0, stream>>>(H, b3, dis, OUT, C34, ts);
        int te = E * C34;
        edge_agg_kernel<<<cdiv(te, BLK), BLK, 0, stream>>>(H, dis, src, dst, OUT, C34, te);
    }
}
